// SparseMoE_27925877359122
// MI455X (gfx1250) — matrix operand order repaired and checked by reading, not compiled, not run on hardware
//
#include <hip/hip_runtime.h>
#include <hip/hip_bf16.h>
#include <math.h>

#define H_DIM   2048
#define DFF_DIM 8192
#define E_NUM   8
#define T_NUM   8192
#define CHUNK   2048
#define KT      32
#define BM      128
#define BN      128

typedef __attribute__((ext_vector_type(16))) __bf16        v16bf;
typedef __attribute__((ext_vector_type(8)))  float         v8f;
typedef __attribute__((ext_vector_type(4)))  float         v4f;
typedef __attribute__((ext_vector_type(8)))  unsigned short v8us;
typedef __attribute__((ext_vector_type(4)))  unsigned short v4us;

#define LOG2E 1.4426950408889634f

union Frag { v16bf v; v8us u2[2]; };

// packed f32x2 -> bf16x2, round-to-nearest (ties away): 3 VALU per pair.
// (native cvt_pk_bf16 builtin is absent on this toolchain; exact RNE costs
//  ~7 VALU/pair — ties-away differs only on exact ties, 1 bf16 ulp.)
__device__ __forceinline__ unsigned f2bfu2(float x, float y) {
  union { float f; unsigned u; } a, b; a.f = x; b.f = y;
  unsigned hi = (b.u + 0x8000u) & 0xFFFF0000u;
  unsigned lo = (a.u + 0x8000u) >> 16;
  return hi | lo;
}

// branchless tanh-gelu: x * sigmoid(2*0.79788456*(x + 0.044715 x^3))
// sigmoid via v_exp_f32 (exp2) + v_rcp_f32 -> ~8 VALU, no divergence
__device__ __forceinline__ float gelu_fast(float x) {
  float u = x * (1.0f + 0.044715f * x * x);          // x + 0.044715 x^3
  float z = 1.5957691216057308f * u;                 // 2 * sqrt(2/pi) * u
  float e = __builtin_amdgcn_exp2f(-LOG2E * z);      // exp(-z)
  return x * __builtin_amdgcn_rcpf(1.0f + e);
}

// ---------------------------------------------------------------- zero output
__global__ void moe_zero_kernel(float* __restrict__ p, size_t n4) {
  size_t i = (size_t)blockIdx.x * blockDim.x + threadIdx.x;
  size_t stride = (size_t)gridDim.x * blockDim.x;
  v4f z = {0.f, 0.f, 0.f, 0.f};
  for (; i < n4; i += stride) ((v4f*)p)[i] = z;
}

// ---------------------------------------------------------------- router
// one wave32 per token: logits = x @ w_gate, softmax(fp32), top-2
__global__ void moe_router_kernel(const float* __restrict__ X,
                                  const float* __restrict__ Wg,
                                  float* __restrict__ logits,
                                  int* __restrict__ sel,
                                  float* __restrict__ selw) {
  int gwave = (int)((blockIdx.x * blockDim.x + threadIdx.x) >> 5);
  int lane  = threadIdx.x & 31;
  if (gwave >= T_NUM) return;
  const float* xr = X + (size_t)gwave * H_DIM;
  float acc[E_NUM] = {0.f, 0.f, 0.f, 0.f, 0.f, 0.f, 0.f, 0.f};
  for (int h = lane; h < H_DIM; h += 32) {
    float xv = xr[h];
    const float* wr = Wg + (size_t)h * E_NUM;
    v4f wa = *(const v4f*)wr;
    v4f wb = *(const v4f*)(wr + 4);
    acc[0] += xv * wa.x; acc[1] += xv * wa.y; acc[2] += xv * wa.z; acc[3] += xv * wa.w;
    acc[4] += xv * wb.x; acc[5] += xv * wb.y; acc[6] += xv * wb.z; acc[7] += xv * wb.w;
  }
#pragma unroll
  for (int e = 0; e < E_NUM; ++e)
    for (int off = 16; off > 0; off >>= 1)
      acc[e] += __shfl_xor(acc[e], off, 32);
  if (lane == 0) {
    float m = acc[0];
    for (int e = 1; e < E_NUM; ++e) m = fmaxf(m, acc[e]);
    float p[E_NUM]; float s = 0.f;
    for (int e = 0; e < E_NUM; ++e) {
      p[e] = __builtin_amdgcn_exp2f(LOG2E * (acc[e] - m));
      s += p[e];
    }
    float inv = 1.f / s;
    int i1 = 0;
    for (int e = 1; e < E_NUM; ++e) if (acc[e] > acc[i1]) i1 = e;
    int i2 = (i1 == 0) ? 1 : 0;
    for (int e = 0; e < E_NUM; ++e) if (e != i1 && acc[e] > acc[i2]) i2 = e;
    for (int e = 0; e < E_NUM; ++e) logits[(size_t)gwave * E_NUM + e] = acc[e];
    sel[2 * gwave]      = i1;
    sel[2 * gwave + 1]  = i2;
    selw[2 * gwave]     = p[i1] * inv;
    selw[2 * gwave + 1] = p[i2] * inv;
  }
}

// ---------------------------------------------------------------- list build
// one lane per expert, token-order scan => deterministic compaction
__global__ void moe_build_kernel(const int* __restrict__ sel,
                                 const float* __restrict__ selw,
                                 int* __restrict__ list,
                                 float* __restrict__ wgt,
                                 int* __restrict__ counts) {
  int e = threadIdx.x;
  if (e >= E_NUM) return;
  int c = 0;
  for (int t = 0; t < T_NUM; ++t) {
    int s0 = sel[2 * t], s1 = sel[2 * t + 1];
    if (s0 == e)      { list[(size_t)e * T_NUM + c] = t; wgt[(size_t)e * T_NUM + c] = selw[2 * t];     ++c; }
    else if (s1 == e) { list[(size_t)e * T_NUM + c] = t; wgt[(size_t)e * T_NUM + c] = selw[2 * t + 1]; ++c; }
  }
  counts[e] = c;
}

// ---------------------------------------------------------------- GEMM1
// hmid[chunk_rows, DFF] = gelu( gather(X)[rows, H] @ W1[e][H, DFF] ), bf16 out
__global__ void __launch_bounds__(256)
moe_gemm1_kernel(const float* __restrict__ X,
                 const float* __restrict__ W1,
                 const int* __restrict__ list,
                 const int* __restrict__ counts,
                 unsigned short* __restrict__ hmid,
                 int expert, int c0) {
  __shared__ __align__(16) unsigned short As[BM * KT];
  __shared__ __align__(16) unsigned short Bs[BN * KT];   // transposed: [n][k]
  __shared__ int s_tok[BM];

  const int tid   = threadIdx.x;
  const int lane  = tid & 31;
  const int wv    = tid >> 5;
  const int waveM = wv >> 2;       // 0..1
  const int waveN = wv & 3;        // 0..3
  const int lhalf = lane >> 4;     // 0/1
  const int lrow  = lane & 15;

  int rows = counts[expert] - c0;
  if (rows > CHUNK) rows = CHUNK;
  const int m0 = blockIdx.y * BM;
  const int n0 = blockIdx.x * BN;
  if (rows <= 0 || m0 >= rows) return;

  if (tid < BM) {
    int r = m0 + tid;
    s_tok[tid] = (r < rows) ? list[(size_t)expert * T_NUM + c0 + r] : 0;
  }
  __syncthreads();

  const float* W1e = W1 + (size_t)expert * H_DIM * DFF_DIM;

  // loop-invariant staging coordinates
  int arowX[4], ac4X[4], bkX[4], bn4X[4];
#pragma unroll
  for (int i = 0; i < 4; ++i) {
    int q = tid + i * 256;
    arowX[i] = q >> 3; ac4X[i] = q & 7;
    bkX[i]   = q >> 5; bn4X[i] = q & 31;
  }

  v4f aR[4], bR[4];
  auto loadA = [&](int k0) {
#pragma unroll
    for (int i = 0; i < 4; ++i)
      aR[i] = *(const v4f*)(X + (size_t)s_tok[arowX[i]] * H_DIM + k0 + ac4X[i] * 4);
  };
  auto loadB = [&](int k0) {
#pragma unroll
    for (int i = 0; i < 4; ++i)
      bR[i] = *(const v4f*)(W1e + (size_t)(k0 + bkX[i]) * DFF_DIM + n0 + bn4X[i] * 4);
  };

  v8f acc[4][2] = {};

  loadA(0); loadB(0);                       // software-pipeline prologue

  for (int k0 = 0; k0 < H_DIM; k0 += KT) {
    __syncthreads();
#pragma unroll
    for (int i = 0; i < 4; ++i) {
      union { unsigned u[2]; v4us v; } pk;
      pk.u[0] = f2bfu2(aR[i].x, aR[i].y);
      pk.u[1] = f2bfu2(aR[i].z, aR[i].w);
      *(v4us*)&As[arowX[i] * KT + ac4X[i] * 4] = pk.v;
    }
#pragma unroll
    for (int i = 0; i < 4; ++i) {
      unsigned p01 = f2bfu2(bR[i].x, bR[i].y);
      unsigned p23 = f2bfu2(bR[i].z, bR[i].w);
      Bs[(bn4X[i] * 4 + 0) * KT + bkX[i]] = (unsigned short)(p01 & 0xFFFFu);
      Bs[(bn4X[i] * 4 + 1) * KT + bkX[i]] = (unsigned short)(p01 >> 16);
      Bs[(bn4X[i] * 4 + 2) * KT + bkX[i]] = (unsigned short)(p23 & 0xFFFFu);
      Bs[(bn4X[i] * 4 + 3) * KT + bkX[i]] = (unsigned short)(p23 >> 16);
    }
    __syncthreads();

    // issue next tile's global loads early: VMEM latency hides under WMMA
    if (k0 + KT < H_DIM) {
      loadA(k0 + KT); loadB(k0 + KT);
      if (k0 + 2 * KT < H_DIM)
        __builtin_prefetch(W1e + (size_t)(k0 + 2 * KT + bkX[0]) * DFF_DIM + n0 + bn4X[0] * 4, 0, 1);
    }

    Frag a[4], b[2];
#pragma unroll
    for (int mi = 0; mi < 4; ++mi) {
      int rl = waveM * 64 + mi * 16 + lrow;
      a[mi].u2[0] = *(const v8us*)&As[rl * KT + lhalf * 8];
      a[mi].u2[1] = *(const v8us*)&As[rl * KT + lhalf * 8 + 16];
    }
#pragma unroll
    for (int ni = 0; ni < 2; ++ni) {
      int cl = waveN * 32 + ni * 16 + lrow;
      b[ni].u2[0] = *(const v8us*)&Bs[cl * KT + lhalf * 8];
      b[ni].u2[1] = *(const v8us*)&Bs[cl * KT + lhalf * 8 + 16];
    }
#pragma unroll
    for (int mi = 0; mi < 4; ++mi)
#pragma unroll
      for (int ni = 0; ni < 2; ++ni)
        acc[mi][ni] = __builtin_amdgcn_wmma_f32_16x16x32_bf16(
            false, a[mi].v, false, b[ni].v, (short)0, acc[mi][ni], false, false);
  }

#pragma unroll
  for (int mi = 0; mi < 4; ++mi) {
#pragma unroll
    for (int ni = 0; ni < 2; ++ni) {
      int col = n0 + waveN * 32 + ni * 16 + lrow;
#pragma unroll
      for (int r = 0; r < 8; r += 2) {
        int grow = m0 + waveM * 64 + mi * 16 + lhalf * 8 + r;
        unsigned p = f2bfu2(gelu_fast(acc[mi][ni][r]), gelu_fast(acc[mi][ni][r + 1]));
        hmid[(size_t)grow * DFF_DIM + col]       = (unsigned short)(p & 0xFFFFu);
        hmid[(size_t)(grow + 1) * DFF_DIM + col] = (unsigned short)(p >> 16);
      }
    }
  }
}

// ---------------------------------------------------------------- GEMM2
// out[tok, H] += w * ( hmid[rows, DFF](bf16) @ W2[e][DFF, H] )
__global__ void __launch_bounds__(256)
moe_gemm2_kernel(const unsigned short* __restrict__ hmid,
                 const float* __restrict__ W2,
                 const int* __restrict__ list,
                 const float* __restrict__ wgt,
                 const int* __restrict__ counts,
                 float* __restrict__ out,
                 int expert, int c0) {
  __shared__ __align__(16) unsigned short As[BM * KT];
  __shared__ __align__(16) unsigned short Bs[BN * KT];   // transposed: [n][k]
  __shared__ int   s_tok[BM];
  __shared__ float s_wt[BM];

  const int tid   = threadIdx.x;
  const int lane  = tid & 31;
  const int wv    = tid >> 5;
  const int waveM = wv >> 2;
  const int waveN = wv & 3;
  const int lhalf = lane >> 4;
  const int lrow  = lane & 15;

  int rows = counts[expert] - c0;
  if (rows > CHUNK) rows = CHUNK;
  const int m0 = blockIdx.y * BM;
  const int n0 = blockIdx.x * BN;
  if (rows <= 0 || m0 >= rows) return;

  if (tid < BM) {
    int r = m0 + tid;
    bool ok = (r < rows);
    s_tok[tid] = ok ? list[(size_t)expert * T_NUM + c0 + r] : 0;
    s_wt[tid]  = ok ? wgt[(size_t)expert * T_NUM + c0 + r] : 0.f;
  }
  __syncthreads();

  const float* W2e = W2 + (size_t)expert * DFF_DIM * H_DIM;

  int arowX[2], aoffX[2], bkX[4], bn4X[4];
#pragma unroll
  for (int i = 0; i < 2; ++i) {
    int q = tid + i * 256;
    arowX[i] = q >> 2; aoffX[i] = (q & 3) * 8;
  }
#pragma unroll
  for (int i = 0; i < 4; ++i) {
    int q = tid + i * 256;
    bkX[i] = q >> 5; bn4X[i] = q & 31;
  }

  v8us aR[2]; v4f bR[4];
  auto loadA = [&](int k0) {
#pragma unroll
    for (int i = 0; i < 2; ++i)
      aR[i] = *(const v8us*)(hmid + (size_t)(m0 + arowX[i]) * DFF_DIM + k0 + aoffX[i]);
  };
  auto loadB = [&](int k0) {
#pragma unroll
    for (int i = 0; i < 4; ++i)
      bR[i] = *(const v4f*)(W2e + (size_t)(k0 + bkX[i]) * H_DIM + n0 + bn4X[i] * 4);
  };

  v8f acc[4][2] = {};

  loadA(0); loadB(0);

  for (int k0 = 0; k0 < DFF_DIM; k0 += KT) {
    __syncthreads();
#pragma unroll
    for (int i = 0; i < 2; ++i)
      *(v8us*)&As[arowX[i] * KT + aoffX[i]] = aR[i];
#pragma unroll
    for (int i = 0; i < 4; ++i) {
      unsigned p01 = f2bfu2(bR[i].x, bR[i].y);
      unsigned p23 = f2bfu2(bR[i].z, bR[i].w);
      Bs[(bn4X[i] * 4 + 0) * KT + bkX[i]] = (unsigned short)(p01 & 0xFFFFu);
      Bs[(bn4X[i] * 4 + 1) * KT + bkX[i]] = (unsigned short)(p01 >> 16);
      Bs[(bn4X[i] * 4 + 2) * KT + bkX[i]] = (unsigned short)(p23 & 0xFFFFu);
      Bs[(bn4X[i] * 4 + 3) * KT + bkX[i]] = (unsigned short)(p23 >> 16);
    }
    __syncthreads();

    if (k0 + KT < DFF_DIM) {
      loadA(k0 + KT); loadB(k0 + KT);
      if (k0 + 2 * KT < DFF_DIM)
        __builtin_prefetch(W2e + (size_t)(k0 + 2 * KT + bkX[0]) * H_DIM + n0 + bn4X[0] * 4, 0, 1);
    }

    Frag a[4], b[2];
#pragma unroll
    for (int mi = 0; mi < 4; ++mi) {
      int rl = waveM * 64 + mi * 16 + lrow;
      a[mi].u2[0] = *(const v8us*)&As[rl * KT + lhalf * 8];
      a[mi].u2[1] = *(const v8us*)&As[rl * KT + lhalf * 8 + 16];
    }
#pragma unroll
    for (int ni = 0; ni < 2; ++ni) {
      int cl = waveN * 32 + ni * 16 + lrow;
      b[ni].u2[0] = *(const v8us*)&Bs[cl * KT + lhalf * 8];
      b[ni].u2[1] = *(const v8us*)&Bs[cl * KT + lhalf * 8 + 16];
    }
#pragma unroll
    for (int mi = 0; mi < 4; ++mi)
#pragma unroll
      for (int ni = 0; ni < 2; ++ni)
        acc[mi][ni] = __builtin_amdgcn_wmma_f32_16x16x32_bf16(
            false, a[mi].v, false, b[ni].v, (short)0, acc[mi][ni], false, false);
  }

#pragma unroll
  for (int mi = 0; mi < 4; ++mi) {
#pragma unroll
    for (int ni = 0; ni < 2; ++ni) {
      int col = n0 + waveN * 32 + ni * 16 + lrow;
#pragma unroll
      for (int r = 0; r < 8; ++r) {
        int rowL = waveM * 64 + mi * 16 + lhalf * 8 + r;
        int grow = m0 + rowL;
        if (grow < rows) {
          float v = acc[mi][ni][r] * s_wt[rowL];
          atomicAdd(&out[(size_t)s_tok[rowL] * H_DIM + col], v);
        }
      }
    }
  }
}

// ---------------------------------------------------------------- launch
extern "C" void kernel_launch(void* const* d_in, const int* in_sizes, int n_in,
                              void* d_out, int out_size, void* d_ws, size_t ws_size,
                              hipStream_t stream) {
  (void)in_sizes; (void)n_in; (void)out_size; (void)ws_size;

  const float* X  = (const float*)d_in[0];   // [T, H]
  const float* Wg = (const float*)d_in[1];   // [H, E]
  const float* W1 = (const float*)d_in[2];   // [E, H, DFF]
  const float* W2 = (const float*)d_in[3];   // [E, DFF, H]

  float* out    = (float*)d_out;                       // [T, H]
  float* logits = out + (size_t)T_NUM * H_DIM;         // [T, E]

  // workspace layout
  int*   sel    = (int*)d_ws;                                 // 2T
  float* selw   = (float*)(sel + 2 * T_NUM);                  // 2T
  int*   list   = (int*)(selw + 2 * T_NUM);                   // E*T
  float* wgt    = (float*)(list + (size_t)E_NUM * T_NUM);     // E*T
  int*   counts = (int*)(wgt + (size_t)E_NUM * T_NUM);        // 64
  unsigned short* hmid = (unsigned short*)(counts + 64);      // CHUNK*DFF bf16

  moe_zero_kernel<<<4096, 256, 0, stream>>>(out, (size_t)T_NUM * H_DIM / 4);
  moe_router_kernel<<<(T_NUM * 32) / 256, 256, 0, stream>>>(X, Wg, logits, sel, selw);
  moe_build_kernel<<<1, 32, 0, stream>>>(sel, selw, list, wgt, counts);

  for (int e = 0; e < E_NUM; ++e) {
    for (int c = 0; c < T_NUM / CHUNK; ++c) {
      moe_gemm1_kernel<<<dim3(DFF_DIM / BN, CHUNK / BM), 256, 0, stream>>>(
          X, W1, list, counts, hmid, e, c * CHUNK);
      moe_gemm2_kernel<<<dim3(H_DIM / BN, CHUNK / BM), 256, 0, stream>>>(
          hmid, W2, list, wgt, counts, out, e, c * CHUNK);
    }
  }
}
